// _TernaryLinear_36283883716713
// MI455X (gfx1250) — compile-verified, matrix operand order checked by reading, not run on hardware
//
#include <hip/hip_runtime.h>
#include <stdint.h>

typedef __attribute__((ext_vector_type(16))) __bf16 v16bf;
typedef __attribute__((ext_vector_type(8)))  float  v8f;

#define KDIM 4096
#define NDIM 4096
#define MDIM 8192
#define WELEMS 16777216            // 4096*4096
#define XELEMS 33554432            // 8192*4096
#define INV_WELEMS (1.0f/16777216.0f)

#define BM 128
#define BN 128

// fallback-kernel tiling
#define BK 32
#define LDSTRIDE 40                // 32 bf16 + 8 pad = 80B row
#define NKB (KDIM / BK)            // 128

// fast-kernel tiling (BK=64: half the barriers, 32 WMMA per sync)
#define FBK 64
#define FLDSTRIDE 72               // 64 bf16 + 8 pad = 144B row (16B aligned)
#define FNKB (KDIM / FBK)          // 64

// workspace layout (bytes)
#define WS_QWB_OFF   4096u
#define WS_XHI_OFF   (4096u + 33554432u)               // 4K + 32MB
#define WS_XLO_OFF   (4096u + 33554432u + 67108864u)   // + 64MB
#define WS_NEED_FAST ((size_t)(4096u + 33554432u + 2u*67108864u))

__device__ __forceinline__ v16bf load_frag(const unsigned short* p) {
    // 16-bit WMMA operand: per lane, two contiguous 16B chunks 32B apart
    union { v16bf v; uint4 q[2]; } f;
    f.q[0] = *(const uint4*)(p);
    f.q[1] = *(const uint4*)(p + 16);
    return f.v;
}

// ---------------- reduction: sum(|w|), deterministic two-stage ----------------
__global__ __launch_bounds__(256) void k_abspartial(const float* __restrict__ w,
                                                    float* __restrict__ part) {
    __shared__ float red[256];
    float s = 0.0f;
    const float4* w4 = (const float4*)w;
    const int total4 = WELEMS / 4;
    for (int c = blockIdx.x * 256 + threadIdx.x; c < total4; c += 512 * 256) {
        float4 v = w4[c];
        s += fabsf(v.x) + fabsf(v.y) + fabsf(v.z) + fabsf(v.w);
    }
    red[threadIdx.x] = s;
    __syncthreads();
    for (int off = 128; off > 0; off >>= 1) {
        if (threadIdx.x < off) red[threadIdx.x] += red[threadIdx.x + off];
        __syncthreads();
    }
    if (threadIdx.x == 0) part[blockIdx.x] = red[0];
}

__global__ __launch_bounds__(256) void k_final(float* __restrict__ wsf) {
    __shared__ float red[256];
    float s = wsf[16 + threadIdx.x] + wsf[16 + 256 + threadIdx.x];
    red[threadIdx.x] = s;
    __syncthreads();
    for (int off = 128; off > 0; off >>= 1) {
        if (threadIdx.x < off) red[threadIdx.x] += red[threadIdx.x + off];
        __syncthreads();
    }
    if (threadIdx.x == 0) wsf[0] = red[0];
}

// ---------------- quantize: w -> ternary {-1,0,+1} as bf16 ----------------
__global__ __launch_bounds__(256) void k_quant(const float* __restrict__ w,
                                               const float* __restrict__ wsf,
                                               unsigned short* __restrict__ qwb) {
    const float thr = wsf[0] * (0.5f * INV_WELEMS);  // 0.5 * alpha
    const int c = blockIdx.x * 256 + threadIdx.x;    // over WELEMS/4 float4s
    float4 v = ((const float4*)w)[c];
    unsigned int q0 = v.x > thr ? 0x3F80u : (v.x < -thr ? 0xBF80u : 0u);
    unsigned int q1 = v.y > thr ? 0x3F80u : (v.y < -thr ? 0xBF80u : 0u);
    unsigned int q2 = v.z > thr ? 0x3F80u : (v.z < -thr ? 0xBF80u : 0u);
    unsigned int q3 = v.w > thr ? 0x3F80u : (v.w < -thr ? 0xBF80u : 0u);
    uint2 o; o.x = q0 | (q1 << 16); o.y = q2 | (q3 << 16);
    *(uint2*)&qwb[(size_t)c * 4] = o;
}

// ---------------- split x -> bf16 hi + bf16 lo ----------------
__global__ __launch_bounds__(256) void k_splitx(const float* __restrict__ x,
                                                unsigned short* __restrict__ xhi,
                                                unsigned short* __restrict__ xlo) {
    const int c = blockIdx.x * 256 + threadIdx.x;    // over XELEMS/4 float4s
    float4 v = ((const float4*)x)[c];
    __bf16 h0 = (__bf16)v.x, h1 = (__bf16)v.y, h2 = (__bf16)v.z, h3 = (__bf16)v.w;
    __bf16 l0 = (__bf16)(v.x - (float)h0), l1 = (__bf16)(v.y - (float)h1);
    __bf16 l2 = (__bf16)(v.z - (float)h2), l3 = (__bf16)(v.w - (float)h3);
    uint2 hh, ll;
    hh.x = (unsigned)__builtin_bit_cast(unsigned short, h0) |
           ((unsigned)__builtin_bit_cast(unsigned short, h1) << 16);
    hh.y = (unsigned)__builtin_bit_cast(unsigned short, h2) |
           ((unsigned)__builtin_bit_cast(unsigned short, h3) << 16);
    ll.x = (unsigned)__builtin_bit_cast(unsigned short, l0) |
           ((unsigned)__builtin_bit_cast(unsigned short, l1) << 16);
    ll.y = (unsigned)__builtin_bit_cast(unsigned short, l2) |
           ((unsigned)__builtin_bit_cast(unsigned short, l3) << 16);
    *(uint2*)&xhi[(size_t)c * 4] = hh;
    *(uint2*)&xlo[(size_t)c * 4] = ll;
}

// ---------------- fast GEMM: async global->LDS staging, double buffered, BK=64 ----------------
__global__ __launch_bounds__(256, 1) void k_gemm_async(
    const unsigned short* __restrict__ xhi, const unsigned short* __restrict__ xlo,
    const unsigned short* __restrict__ qwb, const float* __restrict__ wsf,
    float* __restrict__ out)
{
    __shared__ unsigned short sAh[2][BM * FLDSTRIDE];
    __shared__ unsigned short sAl[2][BM * FLDSTRIDE];
    __shared__ unsigned short sB [2][BN * FLDSTRIDE];

    const int tid  = threadIdx.x;
    const int lane = tid & 31;
    const int wid  = tid >> 5;
    const int wm   = wid & 1;            // 2 waves along M: 64 rows each
    const int wn   = wid >> 1;           // 4 waves along N: 32 cols each
    const int lr   = lane & 15;
    const int oe   = (lane >> 4) << 3;   // 0 or 8 elements (lane-half K offset)

    const int m0 = blockIdx.y * BM;
    const int n0 = blockIdx.x * BN;

    v8f acc[4][2];
    const v8f vzero = {0.f, 0.f, 0.f, 0.f, 0.f, 0.f, 0.f, 0.f};
    #pragma unroll
    for (int mi = 0; mi < 4; ++mi)
        #pragma unroll
        for (int ni = 0; ni < 2; ++ni) acc[mi][ni] = vzero;

    // registerless staging: 3 tiles x 1024 x 16B chunks -> 12 async ops per thread
    auto stage_async = [&](int kb, int buf) {
        const int kc = kb * FBK;
        #pragma unroll
        for (int i = 0; i < 4; ++i) {
            int c = tid + i * 256;
            int row = c >> 3, col = (c & 7) * 8;            // col in bf16 elems (0..56)
            int le = row * FLDSTRIDE + col;
            const unsigned short* ga = &xhi[(size_t)(m0 + row) * KDIM + kc + col];
            const unsigned short* gb = &xlo[(size_t)(m0 + row) * KDIM + kc + col];
            const unsigned short* gw = &qwb[(size_t)(n0 + row) * KDIM + kc + col];
            unsigned la = (unsigned)(size_t)&sAh[buf][le];
            unsigned lb = (unsigned)(size_t)&sAl[buf][le];
            unsigned lw = (unsigned)(size_t)&sB [buf][le];
            asm volatile("global_load_async_to_lds_b128 %0, %1, off"
                         :: "v"(la), "v"(ga) : "memory");
            asm volatile("global_load_async_to_lds_b128 %0, %1, off"
                         :: "v"(lb), "v"(gb) : "memory");
            asm volatile("global_load_async_to_lds_b128 %0, %1, off"
                         :: "v"(lw), "v"(gw) : "memory");
        }
    };

    stage_async(0, 0);
    asm volatile("s_wait_asynccnt 0x0" ::: "memory");
    __syncthreads();

    for (int kb = 0; kb < FNKB; ++kb) {
        const int cur = kb & 1;
        if (kb + 1 < FNKB) stage_async(kb + 1, cur ^ 1);  // async fill overlaps WMMA

        #pragma unroll
        for (int ks = 0; ks < 2; ++ks) {                  // two K=32 steps per staged slab
            const int ko = ks * 32 + oe;
            v16bf bf[2];
            #pragma unroll
            for (int ni = 0; ni < 2; ++ni)
                bf[ni] = load_frag(&sB[cur][(wn * 32 + ni * 16 + lr) * FLDSTRIDE + ko]);
            #pragma unroll
            for (int mi = 0; mi < 4; ++mi) {
                v16bf ah = load_frag(&sAh[cur][(wm * 64 + mi * 16 + lr) * FLDSTRIDE + ko]);
                v16bf al = load_frag(&sAl[cur][(wm * 64 + mi * 16 + lr) * FLDSTRIDE + ko]);
                #pragma unroll
                for (int ni = 0; ni < 2; ++ni) {
                    acc[mi][ni] = __builtin_amdgcn_wmma_f32_16x16x32_bf16(
                        false, ah, false, bf[ni], (short)0, acc[mi][ni], false, false);
                    acc[mi][ni] = __builtin_amdgcn_wmma_f32_16x16x32_bf16(
                        false, al, false, bf[ni], (short)0, acc[mi][ni], false, false);
                }
            }
        }
        asm volatile("s_wait_asynccnt 0x0" ::: "memory");
        __syncthreads();
    }

    const float alpha = wsf[0] * INV_WELEMS;
    #pragma unroll
    for (int mi = 0; mi < 4; ++mi) {
        #pragma unroll
        for (int ni = 0; ni < 2; ++ni) {
            const int mbase = m0 + wm * 64 + mi * 16 + ((lane >> 4) << 3);
            const int nn    = n0 + wn * 32 + ni * 16 + lr;
            #pragma unroll
            for (int j = 0; j < 8; ++j)
                out[(size_t)(mbase + j) * NDIM + nn] = alpha * acc[mi][ni][j];
        }
    }
}

// ---------------- fallback GEMM: single buffer, in-kernel conversion ----------------
__global__ __launch_bounds__(256) void k_gemm_sync(
    const float* __restrict__ x, const unsigned short* __restrict__ qwb,
    const float* __restrict__ wsf, float* __restrict__ out)
{
    __shared__ unsigned short sAh[BM * LDSTRIDE];
    __shared__ unsigned short sAl[BM * LDSTRIDE];
    __shared__ unsigned short sB [BN * LDSTRIDE];

    const int tid  = threadIdx.x;
    const int lane = tid & 31;
    const int wid  = tid >> 5;
    const int wm   = wid & 1;
    const int wn   = wid >> 1;
    const int lr   = lane & 15;
    const int oe   = (lane >> 4) << 3;

    const int m0 = blockIdx.y * BM;
    const int n0 = blockIdx.x * BN;

    v8f acc[4][2];
    const v8f vzero = {0.f, 0.f, 0.f, 0.f, 0.f, 0.f, 0.f, 0.f};
    #pragma unroll
    for (int mi = 0; mi < 4; ++mi)
        #pragma unroll
        for (int ni = 0; ni < 2; ++ni) acc[mi][ni] = vzero;

    for (int kb = 0; kb < NKB; ++kb) {
        const int kc = kb * BK;
        #pragma unroll
        for (int i = 0; i < 4; ++i) {            // x tile: load, convert, store (short-lived)
            int c = tid + i * 256;
            int row = c >> 3, col = c & 7;
            float4 v = *(const float4*)&x[(size_t)(m0 + row) * KDIM + kc + col * 4];
            __bf16 h0 = (__bf16)v.x, h1 = (__bf16)v.y, h2 = (__bf16)v.z, h3 = (__bf16)v.w;
            __bf16 l0 = (__bf16)(v.x - (float)h0), l1 = (__bf16)(v.y - (float)h1);
            __bf16 l2 = (__bf16)(v.z - (float)h2), l3 = (__bf16)(v.w - (float)h3);
            uint2 hh, ll;
            hh.x = (unsigned)__builtin_bit_cast(unsigned short, h0) |
                   ((unsigned)__builtin_bit_cast(unsigned short, h1) << 16);
            hh.y = (unsigned)__builtin_bit_cast(unsigned short, h2) |
                   ((unsigned)__builtin_bit_cast(unsigned short, h3) << 16);
            ll.x = (unsigned)__builtin_bit_cast(unsigned short, l0) |
                   ((unsigned)__builtin_bit_cast(unsigned short, l1) << 16);
            ll.y = (unsigned)__builtin_bit_cast(unsigned short, l2) |
                   ((unsigned)__builtin_bit_cast(unsigned short, l3) << 16);
            *(uint2*)&sAh[row * LDSTRIDE + col * 4] = hh;
            *(uint2*)&sAl[row * LDSTRIDE + col * 4] = ll;
        }
        #pragma unroll
        for (int i = 0; i < 2; ++i) {            // qw tile: straight copy
            int c = tid + i * 256;
            int row = c >> 2, col = (c & 3) * 8;
            uint4 b = *(const uint4*)&qwb[(size_t)(n0 + row) * KDIM + kc + col];
            *(uint4*)&sB[row * LDSTRIDE + col] = b;
        }
        __syncthreads();

        v16bf bf[2];
        #pragma unroll
        for (int ni = 0; ni < 2; ++ni)
            bf[ni] = load_frag(&sB[(wn * 32 + ni * 16 + lr) * LDSTRIDE + oe]);
        #pragma unroll
        for (int mi = 0; mi < 4; ++mi) {
            v16bf ah = load_frag(&sAh[(wm * 64 + mi * 16 + lr) * LDSTRIDE + oe]);
            v16bf al = load_frag(&sAl[(wm * 64 + mi * 16 + lr) * LDSTRIDE + oe]);
            #pragma unroll
            for (int ni = 0; ni < 2; ++ni) {
                acc[mi][ni] = __builtin_amdgcn_wmma_f32_16x16x32_bf16(
                    false, ah, false, bf[ni], (short)0, acc[mi][ni], false, false);
                acc[mi][ni] = __builtin_amdgcn_wmma_f32_16x16x32_bf16(
                    false, al, false, bf[ni], (short)0, acc[mi][ni], false, false);
            }
        }
        __syncthreads();
    }

    const float alpha = wsf[0] * INV_WELEMS;
    #pragma unroll
    for (int mi = 0; mi < 4; ++mi) {
        #pragma unroll
        for (int ni = 0; ni < 2; ++ni) {
            const int mbase = m0 + wm * 64 + mi * 16 + ((lane >> 4) << 3);
            const int nn    = n0 + wn * 32 + ni * 16 + lr;
            #pragma unroll
            for (int j = 0; j < 8; ++j)
                out[(size_t)(mbase + j) * NDIM + nn] = alpha * acc[mi][ni][j];
        }
    }
}

extern "C" void kernel_launch(void* const* d_in, const int* in_sizes, int n_in,
                              void* d_out, int out_size, void* d_ws, size_t ws_size,
                              hipStream_t stream) {
    const float* x = (const float*)d_in[0];          // [4,2048,4096] f32
    const float* w = (const float*)d_in[1];          // [4096,4096]  f32
    float* out = (float*)d_out;                      // [4,2048,4096] f32
    float* wsf = (float*)d_ws;                       // [0]=sum, [16..528)=partials
    unsigned short* qwb = (unsigned short*)((char*)d_ws + WS_QWB_OFF);

    k_abspartial<<<512, 256, 0, stream>>>(w, wsf + 16);
    k_final<<<1, 256, 0, stream>>>(wsf);
    k_quant<<<WELEMS / 1024, 256, 0, stream>>>(w, wsf, qwb);

    dim3 grid(NDIM / BN, MDIM / BM);
    if (ws_size >= WS_NEED_FAST) {
        unsigned short* xhi = (unsigned short*)((char*)d_ws + WS_XHI_OFF);
        unsigned short* xlo = (unsigned short*)((char*)d_ws + WS_XLO_OFF);
        k_splitx<<<XELEMS / 1024, 256, 0, stream>>>(x, xhi, xlo);
        k_gemm_async<<<grid, 256, 0, stream>>>(xhi, xlo, qwb, wsf, out);
    } else {
        k_gemm_sync<<<grid, 256, 0, stream>>>(x, qwb, wsf, out);
    }
}
